// FinancialPINN_32976758899303
// MI455X (gfx1250) — matrix operand order repaired and checked by reading, not compiled, not run on hardware
//
#include <hip/hip_runtime.h>

typedef __attribute__((ext_vector_type(16))) _Float16 v16h;
typedef __attribute__((ext_vector_type(8)))  _Float16 v8h;
typedef __attribute__((ext_vector_type(8)))  float    v8f;
typedef __attribute__((ext_vector_type(4)))  int      v4i;

#define HDIM        128
#define KFF         256
#define NLAYERS     8
#define OMEGA0      30.0f
#define ROWS_PER_WG 64            // 4 waves x 16 rows
#define WG_THREADS  128
#define CHUNK       (HDIM * HDIM) // 16384 elements = one staged weight chunk
#define NCHUNKS     26            // 2 in_w + 8 w1 + 8 w2 + 8 skip halves

#if __has_builtin(__builtin_amdgcn_global_load_async_to_lds_b128)
#define HAVE_ASYNC_LDS 1
#else
#define HAVE_ASYNC_LDS 0
#endif

// chunk ids inside the pre-swizzled workspace
#define CH_IN(h)    (h)
#define CH_W1(i)    (2 + (i))
#define CH_W2(i)    (10 + (i))
#define CH_SK(i, h) (18 + ((i) >> 1) * 2 + (h))

__device__ __forceinline__ v8f wmma_f16(v16h a, v16h b, v8f c) {
  return __builtin_amdgcn_wmma_f32_16x16x32_f16(false, a, false, b, (short)0, c,
                                                false, false);
}

__device__ __forceinline__ float fast_tanh(float x) {
#if __has_builtin(__builtin_amdgcn_tanhf)
  return __builtin_amdgcn_tanhf(x);
#elif __has_builtin(__builtin_amdgcn_tanh_f32)
  return __builtin_amdgcn_tanh_f32(x);
#else
  const float e = __expf(2.0f * x);
  return (e - 1.0f) / (e + 1.0f);
#endif
}

// ---- B-fragment swizzle: (k,n) of a 128x128 chunk -> fragment-order offset ----
__device__ __forceinline__ int bfrag_offset(int k, int n) {
  const int frag = ((n >> 4) << 2) | (k >> 5);
  const int lane = (n & 15) | (((k >> 3) & 1) << 4);
  const int h    = (k & 7) | (((k >> 4) & 1) << 3);
  return (((frag << 5) | lane) << 4) | h;
}

// Fallback: stage one fp32 128x128 chunk into LDS with on-the-fly swizzle+convert.
__device__ __forceinline__ void stage_w128_f32(const float* __restrict__ W,
                                               _Float16* __restrict__ wbuf, int tid) {
  __syncthreads();
#pragma unroll 4
  for (int idx = tid; idx < CHUNK; idx += WG_THREADS)
    wbuf[bfrag_offset(idx >> 7, idx & 127)] = (_Float16)W[idx];
  __syncthreads();
}

// Fast path: pre-swizzled f16 chunk in global -> LDS, async if available.
__device__ __forceinline__ void stage_issue(const _Float16* __restrict__ g,
                                            _Float16* __restrict__ dst, int tid) {
#if HAVE_ASYNC_LDS
#pragma unroll
  for (int i = 0; i < CHUNK / 8 / WG_THREADS; ++i) {   // 16 x b128 per thread
    const int e = (i * WG_THREADS + tid) * 8;
    __builtin_amdgcn_global_load_async_to_lds_b128(
        (v4i*)(g + e),
        (__attribute__((address_space(3))) v4i*)(dst + e), 0, 0);
  }
#else
#pragma unroll 4
  for (int i = tid; i < CHUNK / 8; i += WG_THREADS)
    *(v8h*)(dst + i * 8) = *(const v8h*)(g + i * 8);
#endif
}

__device__ __forceinline__ void stage_sync(int fast) {
#if HAVE_ASYNC_LDS
  if (fast) {
#if __has_builtin(__builtin_amdgcn_s_wait_asynccnt)
    __builtin_amdgcn_s_wait_asynccnt(0);
#else
    asm volatile("s_wait_asynccnt 0x0" ::: "memory");
#endif
  }
#endif
  __syncthreads();
}

__device__ __forceinline__ v16h load_bfrag(const _Float16* __restrict__ wbuf,
                                           int ntile, int s, int lane) {
  return *(const v16h*)(wbuf + (((((ntile << 2) | s) << 5) | lane) << 4));
}

// A fragment (16x32, k-slice s) from a row-major f16 activation row.
// rowp points at the lane's row; kb = 8 for lanes 16-31.
__device__ __forceinline__ v16h load_afrag(const _Float16* __restrict__ rowp,
                                           int s, int kb) {
  const _Float16* p = rowp + (s << 5) + kb;
  v8h lo = *(const v8h*)p;
  v8h hi = *(const v8h*)(p + 16);
  return __builtin_shufflevector(lo, hi, 0, 1, 2, 3, 4, 5, 6, 7,
                                 8, 9, 10, 11, 12, 13, 14, 15);
}

// ---------------- prologue: convert+swizzle all weights into d_ws -------------
__global__ __launch_bounds__(256)
void swizzle_weights(const float* __restrict__ in_w, const float* __restrict__ w1,
                     const float* __restrict__ w2, const float* __restrict__ sw,
                     _Float16* __restrict__ ws) {
  const int c = blockIdx.x;            // 0..25
  const float* src;
  if (c < 2)        src = in_w + c * CHUNK;
  else if (c < 10)  src = w1 + (c - 2) * CHUNK;
  else if (c < 18)  src = w2 + (c - 10) * CHUNK;
  else              src = sw + (c - 18) * CHUNK;   // skip chunks are contiguous
  _Float16* dst = ws + c * CHUNK;
  for (int idx = threadIdx.x; idx < CHUNK; idx += 256)
    dst[bfrag_offset(idx >> 7, idx & 127)] = (_Float16)src[idx];
}

// ------------------------------- main kernel ---------------------------------
__global__ __launch_bounds__(WG_THREADS, 1)
void pinn_siren_wmma(const float* __restrict__ S, const float* __restrict__ T,
                     const float* __restrict__ B,
                     const float* __restrict__ in_w, const float* __restrict__ in_b,
                     const float* __restrict__ w1, const float* __restrict__ b1,
                     const float* __restrict__ w2, const float* __restrict__ b2,
                     const float* __restrict__ sw, const float* __restrict__ sb,
                     const float* __restrict__ ow, const float* __restrict__ ob,
                     const _Float16* __restrict__ wsw, int use_ws,
                     float* __restrict__ out, int npts) {
  __shared__ alignas(64) _Float16 wA[CHUNK];              // 32 KB weight buf A
  __shared__ alignas(64) _Float16 wB[CHUNK];              // 32 KB weight buf B
  __shared__ alignas(64) _Float16 ffbuf[4 * 16 * KFF];    // 32 KB fourier feats
  __shared__ alignas(64) _Float16 xstage[4 * 16 * HDIM];  // 16 KB act staging

  const int  tid  = threadIdx.x;
  const int  lane = tid & 31;
  const int  wave = tid >> 5;
  const int  rowbase = blockIdx.x * ROWS_PER_WG + wave * 16;
  const int  mrow = lane & 15;              // A row / C column-in-tile
  const int  rgrp = (lane >> 4) << 3;       // C row offset (0 or 8)
  const int  kb   = (lane >> 4) << 3;       // A K offset (0 or 8)
  const bool fast = (use_ws != 0);
  _Float16* xs  = xstage + wave * 16 * HDIM;
  _Float16* ffw = ffbuf + wave * 16 * KFF;
  const _Float16* xrow  = xs + mrow * HDIM;
  const _Float16* ffrow = ffw + mrow * KFF;

  // ---- Fourier features into LDS (wave-private tile, row-major [16][256]) ----
  int p = rowbase + mrow;
  p = (p < npts) ? p : (npts - 1);
  const float sv = S[p], tv = T[p];
  const int  hi = lane >> 4;                // 0: sin half, 1: cos half
  _Float16*  ffdst = ffw + mrow * KFF + (hi << 7);
#pragma unroll
  for (int c = 0; c < 16; ++c) {
    v8h vv;
#pragma unroll
    for (int j = 0; j < 8; ++j) {
      const int f = c * 8 + j;
      const float proj = sv * B[f] + tv * B[128 + f];
      vv[j] = (_Float16)(hi ? __cosf(proj) : __sinf(proj));
    }
    *(v8h*)(ffdst + c * 8) = vv;
  }
  asm volatile("s_wait_dscnt 0x0" ::: "memory");

  // ---- input layer: x = sin(OMEGA0 * (ff @ in_w + in_b)) ----
  v8f xc[8];
#pragma unroll
  for (int n = 0; n < 8; ++n)
#pragma unroll
    for (int r = 0; r < 8; ++r) xc[n][r] = 0.0f;

  if (fast) stage_issue(wsw + CH_IN(0) * CHUNK, wA, tid);
  stage_sync(fast);

  {  // half 0 on wA ; prefetch half 1 into wB
    if (fast) stage_issue(wsw + CH_IN(1) * CHUNK, wB, tid);
    else      stage_w128_f32(in_w, wA, tid);
    v16h fa[4];
#pragma unroll
    for (int s = 0; s < 4; ++s) fa[s] = load_afrag(ffrow, s, kb);
#pragma unroll
    for (int n = 0; n < 8; ++n)
#pragma unroll
      for (int s = 0; s < 4; ++s)
        xc[n] = wmma_f16(fa[s], load_bfrag(wA, n, s, lane), xc[n]);
    stage_sync(fast);
  }
  {  // half 1 on wB ; prefetch w1[0] into wA
    if (fast) stage_issue(wsw + CH_W1(0) * CHUNK, wA, tid);
    else      stage_w128_f32(in_w + CHUNK, wB, tid);
    v16h fa[4];
#pragma unroll
    for (int s = 0; s < 4; ++s) fa[s] = load_afrag(ffrow, 4 + s, kb);
#pragma unroll
    for (int n = 0; n < 8; ++n)
#pragma unroll
      for (int s = 0; s < 4; ++s)
        xc[n] = wmma_f16(fa[s], load_bfrag(wB, n, s, lane), xc[n]);
    stage_sync(fast);
  }
#pragma unroll
  for (int n = 0; n < 8; ++n) {
    const float bb = in_b[(n << 4) | mrow];
#pragma unroll
    for (int r = 0; r < 8; ++r) xc[n][r] = __sinf(OMEGA0 * (xc[n][r] + bb));
  }

  // ---- residual blocks (wA holds w1[i] at every loop entry) ----
#pragma unroll 1
  for (int i = 0; i < NLAYERS; ++i) {
    // ---- stage 1: h = tanh(x @ w1 + b1), uses wA; prefetch w2 -> wB ----
    if (fast) stage_issue(wsw + CH_W2(i) * CHUNK, wB, tid);
    else      stage_w128_f32(w1 + i * CHUNK, wA, tid);

#pragma unroll
    for (int n = 0; n < 8; ++n)
#pragma unroll
      for (int r = 0; r < 8; ++r)
        xs[(rgrp + r) * HDIM + (n << 4) + mrow] = (_Float16)xc[n][r];
    asm volatile("s_wait_dscnt 0x0" ::: "memory");

    v16h xa[4];
#pragma unroll
    for (int s = 0; s < 4; ++s) xa[s] = load_afrag(xrow, s, kb);

#pragma unroll
    for (int n = 0; n < 8; ++n) {
      const float bb = b1[i * HDIM + (n << 4) + mrow];
      v8f acc;
#pragma unroll
      for (int r = 0; r < 8; ++r) acc[r] = bb;
#pragma unroll
      for (int s = 0; s < 4; ++s)
        acc = wmma_f16(xa[s], load_bfrag(wA, n, s, lane), acc);
#pragma unroll
      for (int r = 0; r < 8; ++r)
        xs[(rgrp + r) * HDIM + (n << 4) + mrow] = (_Float16)fast_tanh(acc[r]);
    }
    stage_sync(fast);

    // ---- stage 2: x += h @ w2 + b2, uses wB; prefetch next chunk -> wA ----
    {
      const int nxt = ((i & 1) == 0) ? CH_SK(i, 0)
                                     : ((i < NLAYERS - 1) ? CH_W1(i + 1) : -1);
      if (fast && nxt >= 0) stage_issue(wsw + nxt * CHUNK, wA, tid);
    }
    if (!fast) stage_w128_f32(w2 + i * CHUNK, wB, tid);

    asm volatile("s_wait_dscnt 0x0" ::: "memory");
    v16h ta[4];
#pragma unroll
    for (int s = 0; s < 4; ++s) ta[s] = load_afrag(xrow, s, kb);

#pragma unroll
    for (int n = 0; n < 8; ++n) {
      const float bb = b2[i * HDIM + (n << 4) + mrow];
#pragma unroll
      for (int r = 0; r < 8; ++r) xc[n][r] += bb;
#pragma unroll
      for (int s = 0; s < 4; ++s)
        xc[n] = wmma_f16(ta[s], load_bfrag(wB, n, s, lane), xc[n]);
    }
    stage_sync(fast);

    // ---- periodic Fourier skip: x += ff @ skip_w + skip_b ----
    if ((i & 1) == 0) {
      {  // skip half 0 on wA; prefetch skip half 1 -> wB
        if (fast) stage_issue(wsw + CH_SK(i, 1) * CHUNK, wB, tid);
        else      stage_w128_f32(sw + ((i >> 1) * 2) * CHUNK, wA, tid);
        v16h fa[4];
#pragma unroll
        for (int s = 0; s < 4; ++s) fa[s] = load_afrag(ffrow, s, kb);
#pragma unroll
        for (int n = 0; n < 8; ++n)
#pragma unroll
          for (int s = 0; s < 4; ++s)
            xc[n] = wmma_f16(fa[s], load_bfrag(wA, n, s, lane), xc[n]);
        stage_sync(fast);
      }
      {  // skip half 1 on wB; prefetch next layer w1 -> wA
        if (fast) stage_issue(wsw + CH_W1(i + 1) * CHUNK, wA, tid);
        else      stage_w128_f32(sw + ((i >> 1) * 2 + 1) * CHUNK, wB, tid);
        v16h fa[4];
#pragma unroll
        for (int s = 0; s < 4; ++s) fa[s] = load_afrag(ffrow, 4 + s, kb);
#pragma unroll
        for (int n = 0; n < 8; ++n)
#pragma unroll
          for (int s = 0; s < 4; ++s)
            xc[n] = wmma_f16(fa[s], load_bfrag(wB, n, s, lane), xc[n]);
        stage_sync(fast);
      }
#pragma unroll
      for (int n = 0; n < 8; ++n) {
        const float bb = sb[(i >> 1) * HDIM + (n << 4) + mrow];
#pragma unroll
        for (int r = 0; r < 8; ++r) xc[n][r] += bb;
      }
    }
  }

  // ---- output head: out = x @ out_w + out_b (reduce over 128 columns) ----
  float part[8];
#pragma unroll
  for (int r = 0; r < 8; ++r) part[r] = 0.0f;
#pragma unroll
  for (int n = 0; n < 8; ++n) {
    const float w = ow[(n << 4) | mrow];
#pragma unroll
    for (int r = 0; r < 8; ++r) part[r] += xc[n][r] * w;
  }
#pragma unroll
  for (int m = 8; m >= 1; m >>= 1)
#pragma unroll
    for (int r = 0; r < 8; ++r) part[r] += __shfl_xor(part[r], m, 32);

  if (mrow == 0) {
    const float o = ob[0];
#pragma unroll
    for (int r = 0; r < 8; ++r) {
      const int row = rowbase + rgrp + r;
      if (row < npts) out[row] = part[r] + o;
    }
  }
}

extern "C" void kernel_launch(void* const* d_in, const int* in_sizes, int n_in,
                              void* d_out, int out_size, void* d_ws, size_t ws_size,
                              hipStream_t stream) {
  const float* S    = (const float*)d_in[0];
  const float* T    = (const float*)d_in[1];
  const float* B    = (const float*)d_in[2];
  const float* in_w = (const float*)d_in[3];
  const float* in_b = (const float*)d_in[4];
  const float* w1   = (const float*)d_in[5];
  const float* b1   = (const float*)d_in[6];
  const float* w2   = (const float*)d_in[7];
  const float* b2   = (const float*)d_in[8];
  const float* sw   = (const float*)d_in[9];
  const float* sb   = (const float*)d_in[10];
  const float* ow   = (const float*)d_in[11];
  const float* ob   = (const float*)d_in[12];
  float* out = (float*)d_out;

  const int    npts   = in_sizes[0];
  const int    blocks = (npts + ROWS_PER_WG - 1) / ROWS_PER_WG;
  const size_t need   = (size_t)NCHUNKS * CHUNK * 2;  // f16 bytes
  const int    use_ws = (d_ws != nullptr && ws_size >= need) ? 1 : 0;
  _Float16*    wsp    = (_Float16*)d_ws;

  if (use_ws)
    hipLaunchKernelGGL(swizzle_weights, dim3(NCHUNKS), dim3(256), 0, stream,
                       in_w, w1, w2, sw, wsp);
  hipLaunchKernelGGL(pinn_siren_wmma, dim3(blocks), dim3(WG_THREADS), 0, stream,
                     S, T, B, in_w, in_b, w1, b1, w2, b2, sw, sb, ow, ob,
                     (const _Float16*)wsp, use_ws, out, npts);
}
